// QuantumMultiHeadAttention_18245021073627
// MI455X (gfx1250) — matrix operand order repaired and checked by reading, not compiled, not run on hardware
//
#include <hip/hip_runtime.h>

// Problem constants (reference: B,S,D,H = 2,2048,1024,16; HD=64)
#define B_  2
#define S_  2048
#define D_  1024
#define H_  16
#define HD_ 64
#define M_  (B_ * S_)   // 4096 rows for the projection GEMMs

typedef __attribute__((ext_vector_type(16))) __bf16 v16bf;
typedef __attribute__((ext_vector_type(8)))  float  v8f;
typedef __attribute__((ext_vector_type(4)))  float  v4f;   // native vector for NT stores
typedef __attribute__((ext_vector_type(4)))  unsigned int v4u;
typedef __attribute__((ext_vector_type(8)))  int    v8i;
typedef __attribute__((ext_vector_type(4)))  int    v4i;

#if defined(__has_builtin)
#  if __has_builtin(__builtin_amdgcn_tensor_load_to_lds)
#    define HAVE_TDM 1
#  else
#    define HAVE_TDM 0
#  endif
#else
#  define HAVE_TDM 0
#endif

#if !HAVE_TDM
#warning "CDNA5 probe: __builtin_amdgcn_tensor_load_to_lds NOT available -> manual staging fallback in out-proj GEMM"
#endif

union Frag16 { v16bf v; unsigned short u[16]; uint2 q2[4]; uint4 q4[2]; };
union Frag8f { v8f v; float f[8]; };

__device__ __forceinline__ unsigned short f2bf(float f) {
  unsigned int u = __float_as_uint(f);
  u += 0x7FFFu + ((u >> 16) & 1u);          // round-to-nearest-even
  return (unsigned short)(u >> 16);
}
__device__ __forceinline__ unsigned int pack2bf(float a, float b) {
  return (unsigned int)f2bf(a) | ((unsigned int)f2bf(b) << 16);
}

__device__ __forceinline__ v8f wmma_bf16(const Frag16& a, const Frag16& b, v8f c) {
  // D = A(16x32 bf16) * B(32x16 bf16) + C(16x16 f32)
  return __builtin_amdgcn_wmma_f32_16x16x32_bf16(false, a.v, false, b.v,
                                                 (short)0, c, false, false);
}

// ---------------------------------------------------------------------------
// TDM: async DMA of a 2-D bf16 tile (rows x 64) global -> LDS, row pitch in
// LDS padded to 68 ushorts via pad_interval=32 dwords / pad_amount=2 dwords.
// ---------------------------------------------------------------------------
__device__ __forceinline__ void tdm_load_tile_bf16(unsigned lds_addr, const void* gptr,
                                                   int rows, int stride_elems) {
#if HAVE_TDM
  unsigned long long ga = (unsigned long long)(uintptr_t)gptr;
  v4u g0;
  g0[0] = 1u;                                          // count=1, user D#
  g0[1] = lds_addr;                                    // LDS byte address
  g0[2] = (unsigned)(ga & 0xFFFFFFFFull);              // global_addr[31:0]
  g0[3] = (unsigned)((ga >> 32) & 0x01FFFFFFull) | (2u << 30);  // [56:32] | type=2
  v8i g1;
  g1[0] = (int)((1u << 16) | (1u << 20) | (4u << 22) | (1u << 25));
          // data_size=2B | pad_enable | pad_interval=32dw | pad_amount=2dw
  g1[1] = (int)(64u << 16);                            // tensor_dim0=64 (lo16)
  g1[2] = (int)(((unsigned)rows & 0xFFFFu) << 16);     // dim0 hi=0 | tensor_dim1 lo16
  g1[3] = (int)((((unsigned)rows >> 16) & 0xFFFFu) | (64u << 16));  // dim1 hi | tile_dim0=64
  g1[4] = (int)((unsigned)rows & 0xFFFFu);             // tile_dim1=rows, tile_dim2=0
  g1[5] = stride_elems;                                // tensor_dim0_stride lo32
  g1[6] = 0; g1[7] = 0;
  v4i z = {0, 0, 0, 0};
#if __clang_major__ >= 23
  v8i z8 = {0, 0, 0, 0, 0, 0, 0, 0};
  __builtin_amdgcn_tensor_load_to_lds(g0, g1, z, z, z8, 0);
#else
  __builtin_amdgcn_tensor_load_to_lds(g0, g1, z, z, 0);
#endif
#endif
}

// ---------------------------------------------------------------------------
// GEMM: C[M,N] = A[M,K] @ W[N,K]^T + bias  (nn.Linear semantics), bf16 WMMA.
// 256 threads = 8 waves; tile 128M x 64N x 64K; register software pipeline.
// CT=true stores C head-transposed as [(b*H+h)*64+hd][s] (for the K tensor),
// using one b128 store per 8 consecutive s values.
// bf16-A variant stages A via the Tensor Data Mover (double-buffered LDS).
// ---------------------------------------------------------------------------
template <bool A_F32, bool C_F32, bool CT = false>
__global__ __launch_bounds__(256)
void gemm_wmma_kernel(const float* __restrict__ Af32,
                      const unsigned short* __restrict__ Abf,
                      const float* __restrict__ W,
                      const float* __restrict__ bias,
                      unsigned short* __restrict__ Cbf,
                      float* __restrict__ Cf32,
                      int M, int N, int K)
{
  constexpr bool TDM_A = (!A_F32) && (HAVE_TDM != 0);
  constexpr int ABUF = TDM_A ? 2 : 1;
  __shared__ unsigned short As[ABUF * 128 * 68]; // rows x 64k, stride 68
  __shared__ unsigned short Bs[64 * 68];         // transposed: Bs[kk][n]

  const int tid  = threadIdx.x;
  const int lane = tid & 31;
  const int wave = tid >> 5;
  const int mrow = lane & 15;
  const int half = lane >> 4;
  const int m_blk = blockIdx.y * 128;
  const int n_blk = blockIdx.x * 64;

  float4 ra_f[8];           // A tile f32 pipeline regs
  uint4  ra_b[4];           // A tile bf16 pipeline regs (non-TDM fallback)
  float4 rb[4];             // W 4x4 micro-block

  const int nb = (tid & 15) * 4, kkb = (tid >> 4) * 4;

  auto load_tiles = [&](int k0) {
    if constexpr (A_F32) {
#pragma unroll
      for (int i = 0; i < 8; i++) {
        int lin = i * 256 + tid;
        int row = lin >> 4, seg = lin & 15;
        ra_f[i] = *(const float4*)(Af32 + (size_t)(m_blk + row) * K + k0 + seg * 4);
      }
    } else if constexpr (!TDM_A) {
#pragma unroll
      for (int i = 0; i < 4; i++) {
        int lin = i * 256 + tid;
        int row = lin >> 3, seg = lin & 7;
        ra_b[i] = *(const uint4*)(Abf + (size_t)(m_blk + row) * K + k0 + seg * 8);
      }
    }
#pragma unroll
    for (int nn = 0; nn < 4; nn++)
      rb[nn] = *(const float4*)(W + (size_t)(n_blk + nb + nn) * K + k0 + kkb);
  };

  auto store_tiles = [&]() {
    if constexpr (A_F32) {
#pragma unroll
      for (int i = 0; i < 8; i++) {
        int lin = i * 256 + tid;
        int row = lin >> 4, seg = lin & 15;
        uint2 p; p.x = pack2bf(ra_f[i].x, ra_f[i].y); p.y = pack2bf(ra_f[i].z, ra_f[i].w);
        *(uint2*)(As + row * 68 + seg * 4) = p;
      }
    } else if constexpr (!TDM_A) {
#pragma unroll
      for (int i = 0; i < 4; i++) {
        int lin = i * 256 + tid;
        int row = lin >> 3, seg = lin & 7;
        uint2* d = (uint2*)(As + row * 68 + seg * 8);
        d[0] = make_uint2(ra_b[i].x, ra_b[i].y);
        d[1] = make_uint2(ra_b[i].z, ra_b[i].w);
      }
    }
    float c0[4] = { rb[0].x, rb[0].y, rb[0].z, rb[0].w };
    float c1[4] = { rb[1].x, rb[1].y, rb[1].z, rb[1].w };
    float c2[4] = { rb[2].x, rb[2].y, rb[2].z, rb[2].w };
    float c3[4] = { rb[3].x, rb[3].y, rb[3].z, rb[3].w };
#pragma unroll
    for (int j = 0; j < 4; j++) {
      uint2 p; p.x = pack2bf(c0[j], c1[j]); p.y = pack2bf(c2[j], c3[j]);
      *(uint2*)(Bs + (kkb + j) * 68 + nb) = p;   // 4x4 micro-transpose of W
    }
  };

  Frag8f acc[4];
#pragma unroll
  for (int i = 0; i < 4; i++)
#pragma unroll
    for (int r = 0; r < 8; r++) acc[i].f[r] = 0.f;

  if constexpr (TDM_A) {
    if (wave == 0)
      tdm_load_tile_bf16((unsigned)(uintptr_t)&As[0],
                         Abf + (size_t)m_blk * K, 128, K);
  }
  load_tiles(0);

  for (int k0 = 0; k0 < K; k0 += 64) {
    const int cur = TDM_A ? ((k0 >> 6) & 1) : 0;
    store_tiles();
    if constexpr (TDM_A) {
      if (wave == 0) asm volatile("s_wait_tensorcnt 0x0" ::: "memory");
    }
    __syncthreads();
    if (k0 + 64 < K) {
      if constexpr (TDM_A) {
        if (wave == 0)
          tdm_load_tile_bf16((unsigned)(uintptr_t)&As[(1 - cur) * 128 * 68],
                             Abf + (size_t)m_blk * K + k0 + 64, 128, K);
      }
      load_tiles(k0 + 64);                 // overlap with compute below
    }
    const unsigned short* Ac = As + cur * (128 * 68);
#pragma unroll
    for (int ks = 0; ks < 2; ks++) {
      Frag16 a;
      const unsigned short* ap = Ac + (wave * 16 + mrow) * 68 + ks * 32;
      const uint2* p0 = (const uint2*)(ap + half * 8);
      a.q2[0] = p0[0]; a.q2[1] = p0[1];
      const uint2* p1 = (const uint2*)(ap + 16 + half * 8);
      a.q2[2] = p1[0]; a.q2[3] = p1[1];
#pragma unroll
      for (int ns = 0; ns < 4; ns++) {
        Frag16 b;
        const unsigned short* bp = Bs + (ks * 32) * 68 + ns * 16 + mrow;
#pragma unroll
        for (int s = 0; s < 16; s++)
          b.u[s] = bp[(s < 8 ? half * 8 + s : 16 + half * 8 + (s - 8)) * 68];
        acc[ns].v = wmma_bf16(a, b, acc[ns].v);
      }
    }
    __syncthreads();
  }

#pragma unroll
  for (int ns = 0; ns < 4; ns++) {
    int ncol = n_blk + ns * 16 + mrow;
    float bv = bias[ncol];
    int mbase = m_blk + wave * 16 + half * 8;
    if constexpr (CT) {
      // head-transposed bf16 store: [(b*H+h)*64+hd][s]; f[0..7] = 8 consecutive s
      int hh = ncol >> 6, hd = ncol & 63;
      int bb = mbase >> 11;              // / S_
      int ss = mbase & (S_ - 1);
      size_t base = ((size_t)(bb * H_ + hh) * HD_ + hd) * S_ + ss;
      uint4 pk;
      pk.x = pack2bf(acc[ns].f[0] + bv, acc[ns].f[1] + bv);
      pk.y = pack2bf(acc[ns].f[2] + bv, acc[ns].f[3] + bv);
      pk.z = pack2bf(acc[ns].f[4] + bv, acc[ns].f[5] + bv);
      pk.w = pack2bf(acc[ns].f[6] + bv, acc[ns].f[7] + bv);
      *(uint4*)(Cbf + base) = pk;
    } else {
#pragma unroll
      for (int r = 0; r < 8; r++) {
        float vv = acc[ns].f[r] + bv;
        size_t off = (size_t)(mbase + r) * N + ncol;
        if constexpr (C_F32) Cf32[off] = vv;
        else                 Cbf[off] = f2bf(vv);
      }
    }
  }
}

// ---------------------------------------------------------------------------
// Attention: one workgroup per (b, h, 16-query tile), 256 threads = 8 waves.
// Scores 16x2048 f32 live in LDS (only possible with CDNA5's 320KB WGP LDS).
// K is pre-transposed per head ([b][h][hd][s]) so B-fragments come straight
// from global memory as 16 strided 16-bit loads per lane -- no LDS staging at all.
// ---------------------------------------------------------------------------
#define SC_STRIDE 2052                       // floats; 2048 + 4 pad
#define SMEM_SC_BYTES  (16 * SC_STRIDE * 4)  // 131328
#define SMEM_ACC_BYTES (16 * 64 * 4)         // 4096
#define ATT_SMEM (SMEM_SC_BYTES + SMEM_ACC_BYTES)

__global__ __launch_bounds__(256)
void attention_kernel(const unsigned short* __restrict__ Qw,
                      const unsigned short* __restrict__ KTw,  // [(b*H+h)*64+hd][S]
                      const unsigned short* __restrict__ Vw,   // [b][s][h][hd]
                      const int* __restrict__ mask,
                      float* __restrict__ attnW,
                      unsigned short* __restrict__ AO)
{
  extern __shared__ char smem[];
  float* sc   = (float*)smem;
  float* accb = (float*)(smem + SMEM_SC_BYTES);

  const int tid  = threadIdx.x;
  const int lane = tid & 31;
  const int wave = tid >> 5;
  const int mrow = lane & 15;
  const int half = lane >> 4;
  const int q0 = blockIdx.x * 16;
  const int h  = blockIdx.y;
  const int b  = blockIdx.z;

  // ---- Q fragments (16x64 = two 16x32 A-frags), reused for all key tiles ----
  Frag16 qa[2];
  {
    const unsigned short* qr = Qw + ((size_t)(b * S_ + q0 + mrow)) * D_ + h * HD_;
#pragma unroll
    for (int c = 0; c < 2; c++) {
      const uint2* p0 = (const uint2*)(qr + c * 32 + half * 8);
      qa[c].q2[0] = p0[0]; qa[c].q2[1] = p0[1];
      const uint2* p1 = (const uint2*)(qr + c * 32 + 16 + half * 8);
      qa[c].q2[2] = p1[0]; qa[c].q2[3] = p1[1];
    }
  }

  // ---- Phase 1: scores = (Q K^T)/8, masked, into LDS ----
  const size_t ktbase = (size_t)(b * H_ + h) * HD_ * S_;
  for (int t = 0; t < 16; t++) {
    int kt = t * 8 + wave;      // each wave owns 16 of the 128 key tiles
    int g0k = kt * 16;
    Frag8f cf;
#pragma unroll
    for (int r = 0; r < 8; r++) cf.f[r] = 0.f;
#pragma unroll
    for (int c = 0; c < 2; c++) {
      Frag16 bf;                // B col = key (g0k+mrow), 16 hd values/lane in table k order
      const unsigned short* kp = KTw + ktbase + (size_t)(c * 32) * S_ + g0k + mrow;
#pragma unroll
      for (int s = 0; s < 16; s++)
        bf.u[s] = kp[(size_t)(s < 8 ? half * 8 + s : 16 + half * 8 + (s - 8)) * S_];
      cf.v = wmma_bf16(qa[c], bf, cf.v);
    }
    int keyg = g0k + mrow;
#pragma unroll
    for (int r = 0; r < 8; r++) {
      int qr = half * 8 + r;
      int mv = mask[(size_t)(q0 + qr) * S_ + keyg];   // (1,1,S,S), L2-resident
      float val = mv ? cf.f[r] * 0.125f : -1e9f;
      sc[qr * SC_STRIDE + keyg] = val;
    }
  }
  __syncthreads();

  // ---- Phase 2: softmax (2 rows/wave), float4 passes; nontemporal f32 prob
  //      stores to d_out; pack bf16 probs in place ----
  float* gatt = attnW + ((size_t)(b * H_ + h) * S_ + q0) * S_;
#pragma unroll
  for (int rr2 = 0; rr2 < 2; rr2++) {
    int rr = wave * 2 + rr2;
    float* row = sc + rr * SC_STRIDE;
    float mx = -3.4e38f;
    for (int i = 0; i < 16; i++) {
      float4 v = *(const float4*)(row + (i * 32 + lane) * 4);
      mx = fmaxf(mx, fmaxf(fmaxf(v.x, v.y), fmaxf(v.z, v.w)));
    }
#pragma unroll
    for (int o = 16; o >= 1; o >>= 1) mx = fmaxf(mx, __shfl_xor(mx, o, 32));
    float sum = 0.f;
    for (int i = 0; i < 16; i++) {
      float4* p = (float4*)(row + (i * 32 + lane) * 4);
      float4 v = *p;
      v.x = __expf(v.x - mx); v.y = __expf(v.y - mx);
      v.z = __expf(v.z - mx); v.w = __expf(v.w - mx);
      *p = v;
      sum += (v.x + v.y) + (v.z + v.w);
    }
#pragma unroll
    for (int o = 16; o >= 1; o >>= 1) sum += __shfl_xor(sum, o, 32);
    float inv = 1.f / sum;
    float* gr = gatt + (size_t)rr * S_;
    unsigned short* pr = (unsigned short*)row;   // in-place bf16 pack
    for (int i = 0; i < 16; i++) {
      int c4 = i * 32 + lane;
      float4 v = *(const float4*)(row + c4 * 4);
      v.x *= inv; v.y *= inv; v.z *= inv; v.w *= inv;
      v4f nv; nv.x = v.x; nv.y = v.y; nv.z = v.z; nv.w = v.w;
      __builtin_nontemporal_store(nv, (v4f*)(gr + c4 * 4));  // 537MB write-once stream
      uint2 pk; pk.x = pack2bf(v.x, v.y); pk.y = pack2bf(v.z, v.w);
      *(uint2*)(pr + c4 * 4) = pk;
    }
  }
  __syncthreads();

  // ---- Phase 3: attn_out tile = P(16x2048 bf16, LDS) @ V(2048x64 bf16) ----
#pragma unroll
  for (int i = 0; i < 4; i++) accb[i * 256 + tid] = 0.f;
  __syncthreads();

  Frag8f oacc[4];
#pragma unroll
  for (int i = 0; i < 4; i++)
#pragma unroll
    for (int r = 0; r < 8; r++) oacc[i].f[r] = 0.f;

  const unsigned short* prow = (const unsigned short*)(sc + mrow * SC_STRIDE);
  for (int kc = 0; kc < 8; kc++) {
    int kb = wave * 256 + kc * 32;     // this wave's key slice
    Frag16 pa;
    const uint2* p0 = (const uint2*)(prow + kb + half * 8);
    pa.q2[0] = p0[0]; pa.q2[1] = p0[1];
    const uint2* p1 = (const uint2*)(prow + kb + 16 + half * 8);
    pa.q2[2] = p1[0]; pa.q2[3] = p1[1];
#pragma unroll
    for (int ns = 0; ns < 4; ns++) {
      Frag16 vb;                       // B[kk][n]: lane = hd col, 16 keys/lane in table k order
      const unsigned short* vp = Vw + ((size_t)(b * S_ + kb)) * D_ + h * HD_ + ns * 16 + mrow;
#pragma unroll
      for (int s = 0; s < 16; s++)
        vb.u[s] = vp[(size_t)(s < 8 ? half * 8 + s : 16 + half * 8 + (s - 8)) * D_];
      oacc[ns].v = wmma_bf16(pa, vb, oacc[ns].v);
    }
  }
  // cross-wave reduction via LDS float atomics (ds_add_f32)
#pragma unroll
  for (int ns = 0; ns < 4; ns++)
#pragma unroll
    for (int r = 0; r < 8; r++)
      atomicAdd(&accb[(half * 8 + r) * 64 + ns * 16 + mrow], oacc[ns].f[r]);
  __syncthreads();

#pragma unroll
  for (int i = 0; i < 4; i++) {
    int idx = i * 256 + tid;
    int row = idx >> 6, col = idx & 63;
    AO[((size_t)(b * S_ + q0 + row)) * D_ + h * HD_ + col] = f2bf(accb[idx]);
  }
}

// ---------------------------------------------------------------------------
extern "C" void kernel_launch(void* const* d_in, const int* in_sizes, int n_in,
                              void* d_out, int out_size, void* d_ws, size_t ws_size,
                              hipStream_t stream)
{
  (void)in_sizes; (void)n_in; (void)out_size; (void)ws_size;
  const float* x  = (const float*)d_in[0];
  const int* mask = (const int*)  d_in[1];
  const float* Wq = (const float*)d_in[2];
  const float* bq = (const float*)d_in[3];
  const float* Wk = (const float*)d_in[4];
  const float* bk = (const float*)d_in[5];
  const float* Wv = (const float*)d_in[6];
  const float* bv = (const float*)d_in[7];
  const float* Wo = (const float*)d_in[8];
  const float* bo = (const float*)d_in[9];

  float* out   = (float*)d_out;
  float* attnW = out + (size_t)B_ * S_ * D_;   // tuple output #2: (B,H,S,S) f32

  // workspace: Q, K^T, V, attn_out as bf16 (4 x 8 MB = 32 MB)
  unsigned short* Qw  = (unsigned short*)d_ws;
  unsigned short* KTw = Qw  + (size_t)B_ * S_ * D_;
  unsigned short* Vw  = KTw + (size_t)B_ * S_ * D_;
  unsigned short* AOw = Vw  + (size_t)B_ * S_ * D_;

  dim3 ggrid(D_ / 64, M_ / 128);
  gemm_wmma_kernel<true, false, false><<<ggrid, 256, 0, stream>>>(x, nullptr, Wq, bq, Qw,  nullptr, M_, D_, D_);
  gemm_wmma_kernel<true, false, true ><<<ggrid, 256, 0, stream>>>(x, nullptr, Wk, bk, KTw, nullptr, M_, D_, D_);
  gemm_wmma_kernel<true, false, false><<<ggrid, 256, 0, stream>>>(x, nullptr, Wv, bv, Vw,  nullptr, M_, D_, D_);

  static_assert(ATT_SMEM <= 320 * 1024, "LDS budget");
  (void)hipFuncSetAttribute((const void*)attention_kernel,
                            hipFuncAttributeMaxDynamicSharedMemorySize, ATT_SMEM);
  attention_kernel<<<dim3(S_ / 16, H_, B_), 256, ATT_SMEM, stream>>>(
      Qw, KTw, Vw, mask, attnW, AOw);

  gemm_wmma_kernel<false, true, false><<<ggrid, 256, 0, stream>>>(nullptr, AOw, Wo, bo, nullptr, out, M_, D_, D_);
}
